// DiTBlock_11089605558486
// MI455X (gfx1250) — matrix operand order repaired and checked by reading, not compiled, not run on hardware
//
#include <hip/hip_runtime.h>
#include <hip/hip_bf16.h>

// ---------------------------------------------------------------------------
// DiT block for MI455X (gfx1250, wave32, WMMA).
//  - bf16 WMMA (16x16x32) for the four dense GEMMs (qkv, out, w1, w2), f32 accumulate
//  - f32 WMMA (16x16x4) for attention (hd=72 = 18*4, no padding needed)
//  - 64x1024 fp32 score tile kept entirely in 280KB dynamic LDS per block
//  - GLOBAL_LOAD_ASYNC_TO_LDS_* (ASYNCcnt) for tile staging
// ---------------------------------------------------------------------------

typedef __attribute__((ext_vector_type(16))) __bf16  v16bf;
typedef __attribute__((ext_vector_type(8)))  float   v8f;
typedef __attribute__((ext_vector_type(2)))  float   v2f;

#define DMODEL 1152
#define DMOD6  6912
#define TSEQ   1024
#define NHEAD  16
#define HDIM   72
#define DFF    4608
#define MROWS  8192   // B*T

__device__ __forceinline__ unsigned short f2bf(float f){
    unsigned int u = __float_as_uint(f);
    u += 0x7FFFu + ((u >> 16) & 1u);           // round-to-nearest-even
    return (unsigned short)(u >> 16);
}

__device__ __forceinline__ float gelu_tanh(float x){
    float x3 = x * x * x;
    return 0.5f * x * (1.0f + tanhf(0.7978845608028654f * (x + 0.044715f * x3)));
}

// --- CDNA5 async global->LDS (tracked with ASYNCcnt) -----------------------
// Generic LDS pointers carry the LDS byte address in their low 32 bits
// (ISA 10.2), so the truncated pointer is the VDST
// lds-address operand.
__device__ __forceinline__ void async_ld_b128(void* lds, const void* g){
    unsigned l = (unsigned)(size_t)lds;
    asm volatile("global_load_async_to_lds_b128 %0, %1, off"
                 :: "v"(l), "v"(g) : "memory");
}
__device__ __forceinline__ void async_ld_b32(void* lds, const void* g){
    unsigned l = (unsigned)(size_t)lds;
    asm volatile("global_load_async_to_lds_b32 %0, %1, off"
                 :: "v"(l), "v"(g) : "memory");
}
__device__ __forceinline__ void wait_async0(){
    asm volatile("s_wait_asynccnt 0x0" ::: "memory");
}

// ---------------------------------------------------------------------------
// mods[b, 6D] = silu(c[b]) @ ada_w + ada_b
// ---------------------------------------------------------------------------
__global__ __launch_bounds__(256) void k_mods(const float* __restrict__ c,
                                              const float* __restrict__ ada_w,
                                              const float* __restrict__ ada_b,
                                              float* __restrict__ mods){
    __shared__ float sc[DMODEL];
    int b = blockIdx.y;
    for (int i = threadIdx.x; i < DMODEL; i += 256){
        float v = c[b * DMODEL + i];
        sc[i] = v / (1.0f + __expf(-v));
    }
    __syncthreads();
    int col = blockIdx.x * 256 + threadIdx.x;
    float sum = ada_b[col];
    for (int d = 0; d < DMODEL; ++d)
        sum = fmaf(sc[d], ada_w[(size_t)d * DMOD6 + col], sum);
    mods[(size_t)b * DMOD6 + col] = sum;
}

// ---------------------------------------------------------------------------
// elementwise fp32 -> bf16 (weights once per launch, attention output)
// ---------------------------------------------------------------------------
__global__ __launch_bounds__(256) void k_f32_to_bf16(const float* __restrict__ in,
                                                     unsigned short* __restrict__ out,
                                                     int n){
    int i  = blockIdx.x * 256 + threadIdx.x;
    int st = gridDim.x * 256;
    for (; i < n; i += st) out[i] = f2bf(in[i]);
}

// ---------------------------------------------------------------------------
// y[row] = modulate(layernorm(x[row]), mods[b,shift], mods[b,scale]) -> bf16
// one 256-thread block per row (D=1152)
// ---------------------------------------------------------------------------
__global__ __launch_bounds__(256) void k_ln_mod(const float* __restrict__ x,
                                                const float* __restrict__ mods,
                                                int shift_off, int scale_off,
                                                unsigned short* __restrict__ y){
    __shared__ float r1[256], r2[256];
    int row = blockIdx.x;
    int b   = row >> 10;                        // T = 1024
    const float* xr = x + (size_t)row * DMODEL;
    float s1 = 0.f, s2 = 0.f;
    for (int i = threadIdx.x; i < DMODEL; i += 256){
        float v = xr[i]; s1 += v; s2 += v * v;
    }
    r1[threadIdx.x] = s1; r2[threadIdx.x] = s2;
    __syncthreads();
    for (int s = 128; s > 0; s >>= 1){
        if (threadIdx.x < s){
            r1[threadIdx.x] += r1[threadIdx.x + s];
            r2[threadIdx.x] += r2[threadIdx.x + s];
        }
        __syncthreads();
    }
    float mu  = r1[0] * (1.0f / DMODEL);
    float var = r2[0] * (1.0f / DMODEL) - mu * mu;
    float rs  = rsqrtf(var + 1e-6f);
    const float* mb = mods + (size_t)b * DMOD6;
    unsigned short* yr = y + (size_t)row * DMODEL;
    for (int i = threadIdx.x; i < DMODEL; i += 256){
        float v = (xr[i] - mu) * rs;
        v = v * (1.0f + mb[scale_off + i]) + mb[shift_off + i];
        yr[i] = f2bf(v);
    }
}

// ---------------------------------------------------------------------------
// bf16 GEMM:  C[M,N] = A[M,K] @ Bw[K,N] + bias, epilogue modes:
//   0: Cf = val (fp32)
//   1: Cb = bf16(gelu(val))
//   2: Cf = resid + gate[b, n] * val      (gate pre-offset into mods)
// 128x128x32 tile, 8 waves: wave = (wm in {0,64}) x (wn in {0,32,64,96})
// A tile staged with global_load_async_to_lds_b128 (no VGPR bounce).
// ---------------------------------------------------------------------------
union FragB16 { uint4 u[2]; v16bf v; };

__global__ __launch_bounds__(256) void k_gemm_bf16(
        const unsigned short* __restrict__ A,
        const unsigned short* __restrict__ Bw,
        const float* __restrict__ bias,
        float* __restrict__ Cf,
        unsigned short* __restrict__ Cb,
        const float* __restrict__ resid,
        const float* __restrict__ gate,
        int N, int K, int mode){
    __shared__ __align__(16) unsigned short As[128][40];  // [m][k], +8 pad
    __shared__ __align__(16) unsigned short Bs[128][40];  // [n][k] transposed

    int tid  = threadIdx.x;
    int lane = tid & 31, wave = tid >> 5;
    int l15  = lane & 15, lhi = lane >> 4;
    int bn0  = blockIdx.x * 128;
    size_t bm0 = (size_t)blockIdx.y * 128;
    int wm = (wave & 1) * 64;
    int wn = (wave >> 1) * 32;

    v8f acc[4][2];
    #pragma unroll
    for (int i = 0; i < 4; ++i)
        #pragma unroll
        for (int j = 0; j < 2; ++j) acc[i][j] = {};

    int arow = tid >> 1, akoff = (tid & 1) * 16;   // A: 2 thr/row, 16 bf16 each
    int bk   = tid >> 3, bn    = (tid & 7) * 16;   // B: 8 thr/k-row, 16 bf16 each

    for (int kt = 0; kt < K; kt += 32){
        // B tile: load to regs (needs transpose on the LDS side)
        const uint4* bp = (const uint4*)(Bw + (size_t)(kt + bk) * N + bn0 + bn);
        uint4 b0 = bp[0], b1 = bp[1];
        if (kt + 32 < K){
            __builtin_prefetch(A + (bm0 + arow) * K + kt + 32 + akoff, 0, 0);
            __builtin_prefetch(Bw + (size_t)(kt + 32 + bk) * N + bn0 + bn, 0, 0);
        }
        __syncthreads();                       // prior iteration's LDS reads done
        // A tile: async DMA straight into LDS (ASYNCcnt)
        const unsigned short* agp = A + (bm0 + arow) * K + kt + akoff;
        async_ld_b128(&As[arow][akoff],     agp);
        async_ld_b128(&As[arow][akoff + 8], agp + 8);
        // B tile: transpose into Bs[n][k]
        unsigned int wv[8] = {b0.x, b0.y, b0.z, b0.w, b1.x, b1.y, b1.z, b1.w};
        #pragma unroll
        for (int j = 0; j < 8; ++j){
            Bs[bn + 2*j    ][bk] = (unsigned short)(wv[j] & 0xffffu);
            Bs[bn + 2*j + 1][bk] = (unsigned short)(wv[j] >> 16);
        }
        wait_async0();                         // this wave's async loads landed
        __syncthreads();                       // ... and everyone else's

        FragB16 af[4], bf[2];
        #pragma unroll
        for (int i = 0; i < 4; ++i){           // A frag: lane<16 K0-15, lane>=16 K16-31
            int r = wm + i * 16 + l15;
            af[i].u[0] = *(const uint4*)&As[r][lhi * 16];
            af[i].u[1] = *(const uint4*)&As[r][lhi * 16 + 8];
        }
        #pragma unroll
        for (int j = 0; j < 2; ++j){           // B frag: lane<16 K0-15 of col n, lane>=16 K16-31
            int n = wn + j * 16 + l15;
            bf[j].u[0] = *(const uint4*)&Bs[n][lhi * 16];
            bf[j].u[1] = *(const uint4*)&Bs[n][lhi * 16 + 8];
        }
        #pragma unroll
        for (int i = 0; i < 4; ++i)
            #pragma unroll
            for (int j = 0; j < 2; ++j)
                acc[i][j] = __builtin_amdgcn_wmma_f32_16x16x32_bf16(
                                false, af[i].v, false, bf[j].v,
                                (short)0, acc[i][j], false, false);
    }

    // epilogue: C layout — lane<16: M=r, N=lane; lane>=16: M=8+r, N=lane-16
    #pragma unroll
    for (int i = 0; i < 4; ++i){
        #pragma unroll
        for (int j = 0; j < 2; ++j){
            int gn = bn0 + wn + j * 16 + l15;
            float bv = bias[gn];
            #pragma unroll
            for (int r = 0; r < 8; ++r){
                size_t gm  = bm0 + wm + i * 16 + lhi * 8 + r;
                size_t idx = gm * (size_t)N + gn;
                float val  = acc[i][j][r] + bv;
                if (mode == 0){
                    Cf[idx] = val;
                } else if (mode == 1){
                    Cb[idx] = f2bf(gelu_tanh(val));
                } else {
                    float g = gate[(gm >> 10) * DMOD6 + gn];
                    Cf[idx] = resid[idx] + g * val;
                }
            }
        }
    }
}

// ---------------------------------------------------------------------------
// Attention, one block per (64 q-rows, head, batch).
// Dynamic LDS: Qs[64][80] + S[64][1032] + red[64][4]*2  ~= 280KB (CDNA5 WGP LDS)
// f32 WMMA 16x16x4: hd=72 -> 18 K-steps; PV: K=1024 -> 256 K-steps.
// ---------------------------------------------------------------------------
__global__ __launch_bounds__(256) void k_attn(const float* __restrict__ qkv,
                                              float* __restrict__ o){
    extern __shared__ float smem[];
    float* Qs   = smem;                    // [64][80]
    float* S    = smem + 64 * 80;          // [64][1032]
    float* red  = S + 64 * 1032;           // [64][4] row max
    float* red2 = red + 64 * 4;            // [64][4] row sum

    int qt = blockIdx.x, h = blockIdx.y, b = blockIdx.z;
    int tid = threadIdx.x, lane = tid & 31, wave = tid >> 5;
    int l15 = lane & 15, lhi = lane >> 4;

    // load 64x72 Q tile via async global->LDS b32 gathers
    size_t baseQ = ((size_t)(b * TSEQ + qt * 64)) * 3456 + (size_t)h * HDIM;
    for (int i = tid; i < 64 * HDIM; i += 256){
        int r = i / HDIM, cc = i - r * HDIM;
        async_ld_b32(&Qs[r * 80 + cc], &qkv[baseQ + (size_t)r * 3456 + cc]);
    }
    wait_async0();
    __syncthreads();

    const float scale = 0.11785113019775793f;   // 1/sqrt(72)

    // ---- phase 1: S = Q @ K^T * scale --------------------------------------
    size_t baseK = ((size_t)b * TSEQ) * 3456 + DMODEL + (size_t)h * HDIM;
    for (int w8 = 0; w8 < 8; ++w8){
        int nt = wave + w8 * 8;                 // 64 n-tiles / 8 waves
        v8f a4[4];
        #pragma unroll
        for (int i = 0; i < 4; ++i) a4[i] = {};
        int tok = nt * 16 + l15;                // B frag: col = K-token
        const float* kp0 = qkv + baseK + (size_t)tok * 3456;
        for (int ks = 0; ks < 18; ++ks){
            int k0 = ks * 4 + lhi * 2;          // lane>=16 holds K pair {2,3}
            v2f bfr; bfr.x = kp0[k0]; bfr.y = kp0[k0 + 1];
            #pragma unroll
            for (int mt = 0; mt < 4; ++mt){
                int m = mt * 16 + l15;
                v2f afr; afr.x = Qs[m * 80 + k0]; afr.y = Qs[m * 80 + k0 + 1];
                a4[mt] = __builtin_amdgcn_wmma_f32_16x16x4_f32(
                             false, afr, false, bfr, (short)0, a4[mt], false, false);
            }
        }
        #pragma unroll
        for (int mt = 0; mt < 4; ++mt){
            int ncol = nt * 16 + l15;
            #pragma unroll
            for (int r = 0; r < 8; ++r){
                int m = mt * 16 + lhi * 8 + r;
                S[m * 1032 + ncol] = a4[mt][r] * scale;
            }
        }
    }
    __syncthreads();

    // ---- phase 2: row softmax in LDS ---------------------------------------
    {
        int row = tid >> 2, sub = tid & 3;      // 4 threads per row
        float* Sr = S + row * 1032;
        float mx = -3.0e38f;
        for (int cc = sub; cc < TSEQ; cc += 4) mx = fmaxf(mx, Sr[cc]);
        red[row * 4 + sub] = mx;
        __syncthreads();
        mx = fmaxf(fmaxf(red[row*4], red[row*4+1]), fmaxf(red[row*4+2], red[row*4+3]));
        float sm = 0.f;
        for (int cc = sub; cc < TSEQ; cc += 4){
            float e = __expf(Sr[cc] - mx); Sr[cc] = e; sm += e;
        }
        red2[row * 4 + sub] = sm;
        __syncthreads();
        float tot = red2[row*4] + red2[row*4+1] + red2[row*4+2] + red2[row*4+3];
        float inv = 1.0f / tot;
        for (int cc = sub; cc < TSEQ; cc += 4) Sr[cc] *= inv;
    }
    __syncthreads();

    // ---- phase 3: O = P @ V -------------------------------------------------
    size_t baseV = ((size_t)b * TSEQ) * 3456 + 2 * DMODEL + (size_t)h * HDIM;
    size_t baseO = ((size_t)(b * TSEQ + qt * 64)) * DMODEL + (size_t)h * HDIM;
    for (int idx = wave; idx < 20; idx += 8){   // 4 m-tiles x 5 n-tiles (72->80)
        int mt = idx & 3, nt = idx >> 2;
        int ncol = nt * 16 + l15;
        bool nvalid = ncol < HDIM;
        const float* vp = qkv + baseV + ncol;
        v8f c = {};
        for (int ks = 0; ks < 256; ++ks){
            int k0 = ks * 4 + lhi * 2;
            v2f bfr;
            bfr.x = nvalid ? vp[(size_t)k0 * 3456]       : 0.0f;
            bfr.y = nvalid ? vp[(size_t)(k0 + 1) * 3456] : 0.0f;
            int m = mt * 16 + l15;
            v2f afr; afr.x = S[m * 1032 + k0]; afr.y = S[m * 1032 + k0 + 1];
            c = __builtin_amdgcn_wmma_f32_16x16x4_f32(
                    false, afr, false, bfr, (short)0, c, false, false);
        }
        if (nvalid){
            #pragma unroll
            for (int r = 0; r < 8; ++r){
                int m = mt * 16 + lhi * 8 + r;
                o[baseO + (size_t)m * DMODEL + ncol] = c[r];
            }
        }
    }
}

// ---------------------------------------------------------------------------
extern "C" void kernel_launch(void* const* d_in, const int* in_sizes, int n_in,
                              void* d_out, int out_size, void* d_ws, size_t ws_size,
                              hipStream_t stream){
    const float* x     = (const float*)d_in[0];
    const float* c     = (const float*)d_in[1];
    const float* qkv_w = (const float*)d_in[2];
    const float* qkv_b = (const float*)d_in[3];
    const float* out_w = (const float*)d_in[4];
    const float* out_b = (const float*)d_in[5];
    const float* w1    = (const float*)d_in[6];
    const float* b1    = (const float*)d_in[7];
    const float* w2    = (const float*)d_in[8];
    const float* b2    = (const float*)d_in[9];
    const float* ada_w = (const float*)d_in[10];
    const float* ada_b = (const float*)d_in[11];
    float* out = (float*)d_out;

    char* ws = (char*)d_ws;
    size_t off = 0;
    auto alloc = [&](size_t bytes) -> char* {
        char* p = ws + off;
        off = (off + bytes + 255) & ~(size_t)255;
        return p;
    };

    float*          mods = (float*)         alloc((size_t)8 * DMOD6 * 4);
    unsigned short* wqkv = (unsigned short*)alloc((size_t)DMODEL * 3 * DMODEL * 2);
    unsigned short* wout = (unsigned short*)alloc((size_t)DMODEL * DMODEL * 2);
    unsigned short* w1b  = (unsigned short*)alloc((size_t)DMODEL * DFF * 2);
    unsigned short* w2b  = (unsigned short*)alloc((size_t)DFF * DMODEL * 2);
    unsigned short* ybf  = (unsigned short*)alloc((size_t)MROWS * DMODEL * 2);
    float*          qkv  = (float*)         alloc((size_t)MROWS * 3 * DMODEL * 4);
    float*          ao   = (float*)         alloc((size_t)MROWS * DMODEL * 4);
    unsigned short* aob  = (unsigned short*)alloc((size_t)MROWS * DMODEL * 2);
    float*          x1   = (float*)         alloc((size_t)MROWS * DMODEL * 4);
    unsigned short* hbf  = (unsigned short*)alloc((size_t)MROWS * DFF * 2);

    // adaLN modulations + weight conversions (independent, same stream)
    k_mods<<<dim3(DMOD6 / 256, 8), 256, 0, stream>>>(c, ada_w, ada_b, mods);
    k_f32_to_bf16<<<2048, 256, 0, stream>>>(qkv_w, wqkv, DMODEL * 3 * DMODEL);
    k_f32_to_bf16<<<1024, 256, 0, stream>>>(out_w, wout, DMODEL * DMODEL);
    k_f32_to_bf16<<<2048, 256, 0, stream>>>(w1, w1b, DMODEL * DFF);
    k_f32_to_bf16<<<2048, 256, 0, stream>>>(w2, w2b, DFF * DMODEL);

    // y1 = modulate(LN(x), m0, m1) -> bf16
    k_ln_mod<<<MROWS, 256, 0, stream>>>(x, mods, 0, DMODEL, ybf);
    // qkv = y1 @ qkv_w + qkv_b  (fp32 out for f32-WMMA attention)
    k_gemm_bf16<<<dim3(27, 64), 256, 0, stream>>>(ybf, wqkv, qkv_b, qkv,
                                                  nullptr, nullptr, nullptr,
                                                  3 * DMODEL, DMODEL, 0);
    // attention (score tile resident in 280KB LDS)
    size_t smem = (size_t)(64 * 80 + 64 * 1032 + 64 * 8) * 4;
    k_attn<<<dim3(TSEQ / 64, NHEAD, 8), 256, smem, stream>>>(qkv, ao);
    k_f32_to_bf16<<<2048, 256, 0, stream>>>(ao, aob, MROWS * DMODEL);
    // x1 = x + m2 * (attn @ out_w + out_b)
    k_gemm_bf16<<<dim3(9, 64), 256, 0, stream>>>(aob, wout, out_b, x1,
                                                 nullptr, x, mods + 2 * DMODEL,
                                                 DMODEL, DMODEL, 2);
    // y2 = modulate(LN(x1), m3, m4) -> bf16
    k_ln_mod<<<MROWS, 256, 0, stream>>>(x1, mods, 3 * DMODEL, 4 * DMODEL, ybf);
    // h = gelu(y2 @ w1 + b1) -> bf16
    k_gemm_bf16<<<dim3(36, 64), 256, 0, stream>>>(ybf, w1b, b1, nullptr,
                                                  hbf, nullptr, nullptr,
                                                  DFF, DMODEL, 1);
    // out = x1 + m5 * (h @ w2 + b2)
    k_gemm_bf16<<<dim3(9, 64), 256, 0, stream>>>(hbf, w2b, b2, out,
                                                 nullptr, x1, mods + 5 * DMODEL,
                                                 DMODEL, DFF, 2);
}
